// DFT_12223476924859
// MI455X (gfx1250) — matrix operand order repaired and checked by reading, not compiled, not run on hardware
//
#include <hip/hip_runtime.h>

// ---------------------------------------------------------------------------
// STFT (spectrogram) as two fused GEMMs on CDNA5 WMMA (gfx1250, wave32).
//   out_real[b,f,t] =  sum_k wcos[f,k] * xpad[b, t*HOP + k]
//   out_mimg[b,f,t] = -sum_k wsin[f,k] * xpad[b, t*HOP + k]
//
// Fast path (ws_size >= 16MB):
//   1) wpack_kernel: one-time f32->f16 pack of wsin/wcos into d_ws.
//   2) stft_wmma_async_kernel<EDGE>: W tiles staged LDS-direct via
//      global_load_async_to_lds_b128 (ASYNCcnt, s_wait_asynccnt); X tiles
//      via float2 loads (interior tiles) or reflect gather (edge tiles),
//      packed with v_cvt_pkrtz. Double-buffered LDS, one barrier per 64-K
//      round, 4 v_wmma_f32_16x16x32_f16 per wave per round.
// Fallback path: fully in-register staging (no workspace needed).
// ---------------------------------------------------------------------------

#define N_FFT    2048
#define HOP      512
#define BATCH    16
#define LENGTH   262144
#define N_FREQ   2048
#define N_FRAMES 513

#define TM 64                  // freq rows per block
#define TN 32                  // frames per block
#define TK 64                  // staged K per round (2 WMMA k-steps)
#define NK_IT (N_FFT / TK)     // 32 rounds
#define PITCH 72               // 64 halfs + 8 pad: 144B rows, 16B-aligned b128
#define N_TTILE 17             // frame tiles total; tiles 1..14 are interior

typedef _Float16 half16 __attribute__((ext_vector_type(16)));
typedef __fp16   fp16x2 __attribute__((ext_vector_type(2)));
typedef float    float8 __attribute__((ext_vector_type(8)));

union Frag { half16 h; unsigned int u[8]; };
union Pk   { fp16x2 h; unsigned int u; };

__device__ __forceinline__ unsigned int pkrtz(float a, float b) {
    Pk p; p.h = __builtin_amdgcn_cvt_pkrtz(a, b);
    return p.u;
}

__device__ __forceinline__ int refl(int idx) {
    idx = (idx < 0) ? -idx : idx;
    return (idx >= LENGTH) ? (2 * (LENGTH - 1) - idx) : idx;
}

// gfx1250 async copy: per-lane 16B global -> LDS, tracked on ASYNCcnt.
__device__ __forceinline__ void async_ld_b128(unsigned lds_off, const void* gaddr) {
    asm volatile("global_load_async_to_lds_b128 %0, %1, off"
                 :: "v"(lds_off), "v"(gaddr) : "memory");
}
__device__ __forceinline__ void wait_async0() {
    asm volatile("s_wait_asynccnt 0" ::: "memory");
}

// ---------------------------------------------------------------------------
// Prepass: pack wsin/wcos (f32) -> f16 rows in workspace.
// ws layout: [0 .. 4M halfs) = wsin16, [4M .. 8M halfs) = wcos16 (row-major).
// ---------------------------------------------------------------------------
__global__ __launch_bounds__(256) void wpack_kernel(
    const float* __restrict__ wsin,
    const float* __restrict__ wcos,
    unsigned int* __restrict__ ws)
{
    const size_t g    = (size_t)blockIdx.x * 256 + threadIdx.x; // 8 halfs each
    const size_t base = g * 8;
    const float4 a0 = *(const float4*)&wsin[base];
    const float4 a1 = *(const float4*)&wsin[base + 4];
    const float4 b0 = *(const float4*)&wcos[base];
    const float4 b1 = *(const float4*)&wcos[base + 4];
    unsigned int* osin = ws;
    unsigned int* ocos = ws + ((size_t)N_FREQ * N_FFT / 2);
    *(uint4*)&osin[base / 2] = make_uint4(pkrtz(a0.x, a0.y), pkrtz(a0.z, a0.w),
                                          pkrtz(a1.x, a1.y), pkrtz(a1.z, a1.w));
    *(uint4*)&ocos[base / 2] = make_uint4(pkrtz(b0.x, b0.y), pkrtz(b0.z, b0.w),
                                          pkrtz(b1.x, b1.y), pkrtz(b1.z, b1.w));
}

// ---------------------------------------------------------------------------
// Fast path: W via async LDS-direct DMA of pre-packed f16.
// EDGE=false: frame tiles 1..14 (no reflection possible -> float2 loads).
// EDGE=true : frame tiles {0, 15, 16} with full reflect gather.
// ---------------------------------------------------------------------------
struct __align__(16) Smem {
    _Float16 Ws[2][TM][PITCH];
    _Float16 Wc[2][TM][PITCH];
    _Float16 Xt[2][TN][PITCH];
};

template <bool EDGE>
__global__ __launch_bounds__(256) void stft_wmma_async_kernel(
    const float*    __restrict__ x,
    const _Float16* __restrict__ wsin16,
    const _Float16* __restrict__ wcos16,
    float* __restrict__ out)
{
    __shared__ Smem sm;

    const int tid  = threadIdx.x;
    const int lane = tid & 31;
    const int wave = tid >> 5;
    const int wy   = wave >> 1;
    const int wx   = wave & 1;

    const int b = blockIdx.z;
    const int f0 = blockIdx.y * TM;
    int t0;
    if (EDGE) {
        const int e = blockIdx.x;                    // 0..2
        t0 = ((e == 0) ? 0 : (e == 1) ? 15 : 16) * TN;
    } else {
        t0 = (blockIdx.x + 1) * TN;                  // tiles 1..14
    }
    const float* __restrict__ xb = x + (size_t)b * LENGTH;

    float8 acc_c = {};
    float8 acc_s = {};

    // Fragment geometry (CDNA5 16-bit WMMA VGPR layouts).
    const int m15  = lane & 15;
    const int ka   = (lane < 16) ? 0 : 8;
    const int kb   = (lane < 16) ? 0 : 16;
    const int arow = wy * 16 + m15;
    const int brow = wx * 16 + m15;

    // W async staging: 64 rows x 4 chunks of 16 halfs = 256 b128 per matrix
    const int awr = tid >> 2;                 // row 0..63
    const int awc = (tid & 3) * 16;           // k chunk start (halfs)
    // X staging: 32 frames x 64 k -> 4 packed dwords per thread
    const int xn  = tid >> 5;                 // base frame step
    const int xc2 = (tid & 31) * 2;           // k pair start

    float xv[8];

    // X loader for one round at k-offset `kk` into xv[].
    auto load_x = [&](int kk) {
#pragma unroll
        for (int i = 0; i < 4; ++i) {
            const int n = xn + i * 8;
            const int base = (t0 + n) * HOP + kk + xc2 - (N_FFT / 2);
            if (EDGE) {
                xv[2 * i]     = xb[refl(base)];
                xv[2 * i + 1] = xb[refl(base + 1)];
            } else {
                const float2 v = *(const float2*)&xb[base];  // 8B aligned
                xv[2 * i]     = v.x;
                xv[2 * i + 1] = v.y;
            }
        }
    };

    // ---- prologue: stage tile 0 into buffer 0 ----
    async_ld_b128((unsigned)(size_t)&sm.Ws[0][awr][awc],
                  &wsin16[(size_t)(f0 + awr) * N_FFT + awc]);
    async_ld_b128((unsigned)(size_t)&sm.Wc[0][awr][awc],
                  &wcos16[(size_t)(f0 + awr) * N_FFT + awc]);
    load_x(0);
#pragma unroll
    for (int i = 0; i < 4; ++i) {
        const int n = xn + i * 8;
        *(unsigned int*)&sm.Xt[0][n][xc2] = pkrtz(xv[2 * i], xv[2 * i + 1]);
    }
    wait_async0();

    int cur = 0;
    for (int it = 0; it < NK_IT; ++it) {
        __syncthreads();   // buffer `cur` fully staged for all waves
        const int nb = cur ^ 1;
        const int kn = ((it + 1 < NK_IT) ? (it + 1) : it) * TK;

        // ---- issue async W DMA for next tile (2 x b128 per thread) ----
        async_ld_b128((unsigned)(size_t)&sm.Ws[nb][awr][awc],
                      &wsin16[(size_t)(f0 + awr) * N_FFT + kn + awc]);
        async_ld_b128((unsigned)(size_t)&sm.Wc[nb][awr][awc],
                      &wcos16[(size_t)(f0 + awr) * N_FFT + kn + awc]);

        // ---- issue X global loads for next tile ----
        load_x(kn);

        // ---- compute: 2 k-steps x 2 matrices from buffer `cur` ----
#pragma unroll
        for (int s = 0; s < 2; ++s) {
            Frag a_s, a_c, bf;
#pragma unroll
            for (int j = 0; j < 8; ++j) {
                const int kpa = s * 32 + 2 * j + ((j >= 4) ? 8 : 0) + ka;
                a_s.u[j] = *(const unsigned int*)&sm.Ws[cur][arow][kpa];
                a_c.u[j] = *(const unsigned int*)&sm.Wc[cur][arow][kpa];
                const int kpb = s * 32 + 2 * j + ((j >= 4) ? 8 : 0) + ka;
                bf.u[j]  = *(const unsigned int*)&sm.Xt[cur][brow][kpb];
            }
            acc_c = __builtin_amdgcn_wmma_f32_16x16x32_f16(
                false, a_c.h, false, bf.h, (short)0, acc_c, false, false);
            acc_s = __builtin_amdgcn_wmma_f32_16x16x32_f16(
                false, a_s.h, false, bf.h, (short)0, acc_s, false, false);
        }

        // ---- pack + store X into the other buffer; drain async DMA ----
#pragma unroll
        for (int i = 0; i < 4; ++i) {
            const int n = xn + i * 8;
            *(unsigned int*)&sm.Xt[nb][n][xc2] = pkrtz(xv[2 * i], xv[2 * i + 1]);
        }
        wait_async0();
        cur = nb;
    }

    // ---- epilogue ----
    const int t = t0 + wx * 16 + m15;
    if (t < N_FRAMES) {
        const size_t half_out = (size_t)BATCH * N_FREQ * N_FRAMES;
        const int fbase = f0 + wy * 16 + ((lane < 16) ? 0 : 8);
#pragma unroll
        for (int r = 0; r < 8; ++r) {
            const size_t o = ((size_t)b * N_FREQ + (size_t)(fbase + r)) * N_FRAMES
                           + (size_t)t;
            out[o]            = acc_c[r];
            out[half_out + o] = -acc_s[r];
        }
    }
}

// ---------------------------------------------------------------------------
// Fallback path (no workspace): in-register W staging.
// ---------------------------------------------------------------------------
__global__ __launch_bounds__(256) void stft_wmma_kernel(
    const float* __restrict__ x,
    const float* __restrict__ wsin,
    const float* __restrict__ wcos,
    float* __restrict__ out)
{
    __shared__ _Float16 sWs[2][TM][PITCH];
    __shared__ _Float16 sWc[2][TM][PITCH];
    __shared__ _Float16 sXt[2][TN][PITCH];

    const int tid  = threadIdx.x;
    const int lane = tid & 31;
    const int wave = tid >> 5;
    const int wy   = wave >> 1;
    const int wx   = wave & 1;

    const int b  = blockIdx.z;
    const int f0 = blockIdx.y * TM;
    const int t0 = blockIdx.x * TN;

    float8 acc_c = {};
    float8 acc_s = {};

    const int m15  = lane & 15;
    const int ka   = (lane < 16) ? 0 : 8;
    const int kb   = (lane < 16) ? 0 : 16;
    const int arow = wy * 16 + m15;
    const int brow = wx * 16 + m15;

    const int wr  = tid >> 4;
    const int wc4 = (tid & 15) * 4;
    const int xn  = tid >> 5;
    const int xc2 = (tid & 31) * 2;

    float4 wsv[4], wcv[4];
    float  xv[8];

#pragma unroll
    for (int i = 0; i < 4; ++i) {
        const int r = wr + i * 16;
        wsv[i] = *(const float4*)&wsin[(size_t)(f0 + r) * N_FFT + wc4];
        wcv[i] = *(const float4*)&wcos[(size_t)(f0 + r) * N_FFT + wc4];
        const int n = xn + i * 8;
        const int base = (t0 + n) * HOP + xc2 - (N_FFT / 2);
        xv[2 * i]     = x[(size_t)b * LENGTH + refl(base)];
        xv[2 * i + 1] = x[(size_t)b * LENGTH + refl(base + 1)];
    }
#pragma unroll
    for (int i = 0; i < 4; ++i) {
        const int r = wr + i * 16;
        *(uint2*)&sWs[0][r][wc4] =
            make_uint2(pkrtz(wsv[i].x, wsv[i].y), pkrtz(wsv[i].z, wsv[i].w));
        *(uint2*)&sWc[0][r][wc4] =
            make_uint2(pkrtz(wcv[i].x, wcv[i].y), pkrtz(wcv[i].z, wcv[i].w));
        const int n = xn + i * 8;
        *(unsigned int*)&sXt[0][n][xc2] = pkrtz(xv[2 * i], xv[2 * i + 1]);
    }

    int cur = 0;
    for (int it = 0; it < NK_IT; ++it) {
        __syncthreads();
        const int kn = ((it + 1 < NK_IT) ? (it + 1) : it) * TK;
#pragma unroll
        for (int i = 0; i < 4; ++i) {
            const int r = wr + i * 16;
            wsv[i] = *(const float4*)&wsin[(size_t)(f0 + r) * N_FFT + kn + wc4];
            wcv[i] = *(const float4*)&wcos[(size_t)(f0 + r) * N_FFT + kn + wc4];
            const int n = xn + i * 8;
            const int base = (t0 + n) * HOP + kn + xc2 - (N_FFT / 2);
            xv[2 * i]     = x[(size_t)b * LENGTH + refl(base)];
            xv[2 * i + 1] = x[(size_t)b * LENGTH + refl(base + 1)];
        }
#pragma unroll
        for (int s = 0; s < 2; ++s) {
            Frag a_s, a_c, bf;
#pragma unroll
            for (int j = 0; j < 8; ++j) {
                const int kpa = s * 32 + 2 * j + ((j >= 4) ? 8 : 0) + ka;
                a_s.u[j] = *(const unsigned int*)&sWs[cur][arow][kpa];
                a_c.u[j] = *(const unsigned int*)&sWc[cur][arow][kpa];
                const int kpb = s * 32 + 2 * j + ((j >= 4) ? 8 : 0) + ka;
                bf.u[j]  = *(const unsigned int*)&sXt[cur][brow][kpb];
            }
            acc_c = __builtin_amdgcn_wmma_f32_16x16x32_f16(
                false, a_c.h, false, bf.h, (short)0, acc_c, false, false);
            acc_s = __builtin_amdgcn_wmma_f32_16x16x32_f16(
                false, a_s.h, false, bf.h, (short)0, acc_s, false, false);
        }
        const int nb = cur ^ 1;
#pragma unroll
        for (int i = 0; i < 4; ++i) {
            const int r = wr + i * 16;
            *(uint2*)&sWs[nb][r][wc4] =
                make_uint2(pkrtz(wsv[i].x, wsv[i].y), pkrtz(wsv[i].z, wsv[i].w));
            *(uint2*)&sWc[nb][r][wc4] =
                make_uint2(pkrtz(wcv[i].x, wcv[i].y), pkrtz(wcv[i].z, wcv[i].w));
            const int n = xn + i * 8;
            *(unsigned int*)&sXt[nb][n][xc2] = pkrtz(xv[2 * i], xv[2 * i + 1]);
        }
        cur = nb;
    }

    const int t = t0 + wx * 16 + m15;
    if (t < N_FRAMES) {
        const size_t half_out = (size_t)BATCH * N_FREQ * N_FRAMES;
        const int fbase = f0 + wy * 16 + ((lane < 16) ? 0 : 8);
#pragma unroll
        for (int r = 0; r < 8; ++r) {
            const size_t o = ((size_t)b * N_FREQ + (size_t)(fbase + r)) * N_FRAMES
                           + (size_t)t;
            out[o]            = acc_c[r];
            out[half_out + o] = -acc_s[r];
        }
    }
}

extern "C" void kernel_launch(void* const* d_in, const int* in_sizes, int n_in,
                              void* d_out, int out_size, void* d_ws, size_t ws_size,
                              hipStream_t stream) {
    (void)in_sizes; (void)n_in; (void)out_size;
    const float* x    = (const float*)d_in[0];
    const float* wsin = (const float*)d_in[1];
    const float* wcos = (const float*)d_in[2];
    float* out = (float*)d_out;

    dim3 block(256);
    const size_t need = 2ull * N_FREQ * N_FFT * sizeof(_Float16);  // 16 MB
    if (d_ws != nullptr && ws_size >= need) {
        // pack W once (deterministic every call), then async-DMA main kernels
        const size_t total8 = (size_t)N_FREQ * N_FFT / 8;          // threads
        hipLaunchKernelGGL(wpack_kernel, dim3(total8 / 256), dim3(256), 0, stream,
                           wsin, wcos, (unsigned int*)d_ws);
        const _Float16* wsin16 = (const _Float16*)d_ws;
        const _Float16* wcos16 = wsin16 + (size_t)N_FREQ * N_FFT;
        // interior frame tiles 1..14: no reflection, float2 X loads
        hipLaunchKernelGGL((stft_wmma_async_kernel<false>),
                           dim3(N_TTILE - 3, N_FREQ / TM, BATCH), block, 0, stream,
                           x, wsin16, wcos16, out);
        // edge frame tiles {0, 15, 16}: reflect gather
        hipLaunchKernelGGL((stft_wmma_async_kernel<true>),
                           dim3(3, N_FREQ / TM, BATCH), block, 0, stream,
                           x, wsin16, wcos16, out);
    } else {
        hipLaunchKernelGGL(stft_wmma_kernel,
                           dim3(N_TTILE, N_FREQ / TM, BATCH), block, 0, stream,
                           x, wsin, wcos, out);
    }
}
